// MoERWKVBlock_23605140259537
// MI455X (gfx1250) — matrix operand order repaired and checked by reading, not compiled, not run on hardware
//
#include <hip/hip_runtime.h>

// ---------------------------------------------------------------------------
// Types
// ---------------------------------------------------------------------------
typedef __bf16 bf16_t;
typedef __attribute__((ext_vector_type(16))) __bf16 v16bf;
typedef __attribute__((ext_vector_type(8)))  float  v8f;

__device__ __forceinline__ bf16_t f2bf(float f) {
  union { float f; unsigned u; } a; a.f = f;
  unsigned r = a.u + 0x7FFFu + ((a.u >> 16) & 1u);     // round-to-nearest-even
  union { unsigned short u; bf16_t b; } o; o.u = (unsigned short)(r >> 16);
  return o.b;
}
__device__ __forceinline__ float bf2f(bf16_t b) {
  union { unsigned short u; bf16_t b; } i; i.b = b;
  union { unsigned u; float f; } o; o.u = ((unsigned)i.u) << 16;
  return o.f;
}

// ---------------------------------------------------------------------------
// f32 -> bf16 weight conversion
// ---------------------------------------------------------------------------
__global__ __launch_bounds__(256) void cvt_f32_bf16(const float* __restrict__ in,
                                                    bf16_t* __restrict__ out, size_t n) {
  size_t i = (size_t)blockIdx.x * blockDim.x + threadIdx.x;
  size_t stride = (size_t)gridDim.x * blockDim.x;
  for (; i < n; i += stride) out[i] = f2bf(in[i]);
}

// ---------------------------------------------------------------------------
// LayerNorm: one block (256 threads, 8 wave32) per token, bf16 output
// ---------------------------------------------------------------------------
__global__ __launch_bounds__(256) void ln_kernel(const float* __restrict__ x,
                                                 const float* __restrict__ g,
                                                 const float* __restrict__ b,
                                                 bf16_t* __restrict__ out, int D) {
  int t = blockIdx.x;
  const float* row = x + (size_t)t * D;
  float s = 0.f, s2 = 0.f;
  for (int i = threadIdx.x; i < D; i += blockDim.x) { float v = row[i]; s += v; s2 += v * v; }
  #pragma unroll
  for (int o = 16; o > 0; o >>= 1) { s += __shfl_xor(s, o, 32); s2 += __shfl_xor(s2, o, 32); }
  __shared__ float red[2][8];
  int wid = threadIdx.x >> 5, lane = threadIdx.x & 31;
  if (lane == 0) { red[0][wid] = s; red[1][wid] = s2; }
  __syncthreads();
  if (wid == 0) {
    s  = (lane < 8) ? red[0][lane] : 0.f;
    s2 = (lane < 8) ? red[1][lane] : 0.f;
    #pragma unroll
    for (int o = 4; o > 0; o >>= 1) { s += __shfl_xor(s, o, 32); s2 += __shfl_xor(s2, o, 32); }
    if (lane == 0) { red[0][0] = s; red[1][0] = s2; }
  }
  __syncthreads();
  float mean = red[0][0] / D;
  float var  = red[1][0] / D - mean * mean;
  float inv  = rsqrtf(var + 1e-5f);
  bf16_t* orow = out + (size_t)t * D;
  for (int i = threadIdx.x; i < D; i += blockDim.x)
    orow[i] = f2bf((row[i] - mean) * inv * g[i] + b[i]);
}

// ---------------------------------------------------------------------------
// WKV chunked scan (exact log-sum-exp segment combine)
// ---------------------------------------------------------------------------
__global__ __launch_bounds__(256) void wkv_pass1(const float* __restrict__ k,
                                                 const float* __restrict__ v,
                                                 const float* __restrict__ time_decay,
                                                 float* __restrict__ Ac, float* __restrict__ Bc,
                                                 float* __restrict__ Mc,
                                                 int Bn, int S, int D, int L) {
  int gid = blockIdx.x * blockDim.x + threadIdx.x;
  int NC = S / L;
  if (gid >= Bn * D * NC) return;
  int d = gid % D, b = (gid / D) % Bn, c = gid / (D * Bn);
  float w = -__expf(time_decay[d]);
  float A = 0.f, Bv = 0.f, m = -1e38f;
  size_t base = ((size_t)b * S + (size_t)c * L) * D + d;
  for (int i = 0; i < L; ++i) {
    float kt = k[base + (size_t)i * D];
    float vt = v[base + (size_t)i * D];
    float q  = fmaxf(m + w, kt);
    float e1 = __expf(m + w - q);
    float e2 = __expf(kt - q);
    A = e1 * A + e2 * vt; Bv = e1 * Bv + e2; m = q;
  }
  int cidx = (b * D + d) * NC + c;
  Ac[cidx] = A; Bc[cidx] = Bv; Mc[cidx] = m;
}

__global__ __launch_bounds__(256) void wkv_pass2(const float* __restrict__ Ac,
                                                 const float* __restrict__ Bc,
                                                 const float* __restrict__ Mc,
                                                 const float* __restrict__ time_decay,
                                                 float* __restrict__ Sa, float* __restrict__ Sb,
                                                 float* __restrict__ Sp,
                                                 int Bn, int D, int NC, int L) {
  int gid = blockIdx.x * blockDim.x + threadIdx.x;
  if (gid >= Bn * D) return;
  int d = gid % D;
  float w = -__expf(time_decay[d]);
  float aa = 0.f, bb = 0.f, pp = -1e30f;
  int base = gid * NC;
  for (int c = 0; c < NC; ++c) {
    Sa[base + c] = aa; Sb[base + c] = bb; Sp[base + c] = pp;
    float m  = Mc[base + c];
    float q  = fmaxf(pp + (float)L * w, m);
    float e1 = __expf(pp + (float)L * w - q);
    float e2 = __expf(m - q);
    aa = e1 * aa + e2 * Ac[base + c];
    bb = e1 * bb + e2 * Bc[base + c];
    pp = q;
  }
}

__global__ __launch_bounds__(256) void wkv_pass3(const float* __restrict__ k,
                                                 const float* __restrict__ v,
                                                 const float* __restrict__ time_decay,
                                                 const float* __restrict__ time_first,
                                                 const float* __restrict__ Sa,
                                                 const float* __restrict__ Sb,
                                                 const float* __restrict__ Sp,
                                                 float* __restrict__ wkv,
                                                 int Bn, int S, int D, int L) {
  int gid = blockIdx.x * blockDim.x + threadIdx.x;
  int NC = S / L;
  if (gid >= Bn * D * NC) return;
  int d = gid % D, b = (gid / D) % Bn, c = gid / (D * Bn);
  float w = -__expf(time_decay[d]);
  float u = time_first[d];
  int sidx = (b * D + d) * NC + c;
  float aa = Sa[sidx], bb = Sb[sidx], pp = Sp[sidx];
  size_t base = ((size_t)b * S + (size_t)c * L) * D + d;
  for (int i = 0; i < L; ++i) {
    float kt = k[base + (size_t)i * D];
    float vt = v[base + (size_t)i * D];
    float q  = fmaxf(pp, u + kt);
    float e1 = __expf(pp - q);
    float e2 = __expf(u + kt - q);
    wkv[base + (size_t)i * D] = (e1 * aa + e2 * vt) / (e1 * bb + e2);
    float q2  = fmaxf(pp + w, kt);
    float e1b = __expf(pp + w - q2);
    float e2b = __expf(kt - q2);
    aa = e1b * aa + e2b * vt; bb = e1b * bb + e2b; pp = q2;
  }
}

// ---------------------------------------------------------------------------
// Router: logits -> softmax -> top-2 -> gather lists (per-expert token index)
// ---------------------------------------------------------------------------
__global__ __launch_bounds__(256) void router_kernel(const bf16_t* __restrict__ ei,
                                                     const float* __restrict__ rw,
                                                     int D, int maxM,
                                                     int* __restrict__ counts,
                                                     int* __restrict__ gidx,
                                                     float* __restrict__ gw) {
  int t = blockIdx.x;
  const bf16_t* row = ei + (size_t)t * D;
  float acc[8];
  #pragma unroll
  for (int e = 0; e < 8; ++e) acc[e] = 0.f;
  for (int d = threadIdx.x; d < D; d += 256) {
    float v = bf2f(row[d]);
    const float* w = rw + (size_t)d * 8;
    #pragma unroll
    for (int e = 0; e < 8; ++e) acc[e] += v * w[e];
  }
  #pragma unroll
  for (int o = 16; o > 0; o >>= 1)
    #pragma unroll
    for (int e = 0; e < 8; ++e) acc[e] += __shfl_xor(acc[e], o, 32);
  __shared__ float sred[8][8];
  int wid = threadIdx.x >> 5, lane = threadIdx.x & 31;
  if (lane == 0)
    #pragma unroll
    for (int e = 0; e < 8; ++e) sred[wid][e] = acc[e];
  __syncthreads();
  if (threadIdx.x == 0) {
    float l[8];
    #pragma unroll
    for (int e = 0; e < 8; ++e) {
      float s = 0.f;
      #pragma unroll
      for (int w2 = 0; w2 < 8; ++w2) s += sred[w2][e];
      l[e] = s;
    }
    float mx = l[0];
    #pragma unroll
    for (int e = 1; e < 8; ++e) mx = fmaxf(mx, l[e]);
    float p[8]; float se = 0.f;
    #pragma unroll
    for (int e = 0; e < 8; ++e) { p[e] = __expf(l[e] - mx); se += p[e]; }
    #pragma unroll
    for (int e = 0; e < 8; ++e) p[e] /= se;
    int i0 = 0;
    #pragma unroll
    for (int e = 1; e < 8; ++e) if (p[e] > p[i0]) i0 = e;     // ties -> lowest idx
    int i1 = (i0 == 0) ? 1 : 0;
    #pragma unroll
    for (int e = 0; e < 8; ++e) if (e != i0 && p[e] > p[i1]) i1 = e;
    float sum2 = p[i0] + p[i1];
    int s0 = atomicAdd(&counts[i0], 1);
    gidx[i0 * maxM + s0] = t; gw[i0 * maxM + s0] = p[i0] / sum2;
    int s1 = atomicAdd(&counts[i1], 1);
    gidx[i1 * maxM + s1] = t; gw[i1 * maxM + s1] = p[i1] / sum2;
  }
}

// ---------------------------------------------------------------------------
// bf16 WMMA GEMM, 128x128 block tile, 8 wave32, wave tile 64x32 (8 WMMAs/Kstep)
//   A-modes: 0 = bf16 matrix direct
//            1 = shift-mix from token activations (opt. gathered via gidx)
//            2 = elementwise f32 product (r * wkv) -> bf16
//   Epilogues: 0 = f32 store, 1 = sigmoid f32, 2 = relu^2 bf16,
//              3 = + residual, store to two f32 dests, 4 = MoE scatter-acc
// ---------------------------------------------------------------------------
#define BM 128
#define BN 128
#define BKK 32
#define LDSP 40  // BKK + 8 bf16 pad: 80B row stride -> conflict-free frag reads

template <int AM, int EP>
__global__ __launch_bounds__(256) void gemm_bf16_wmma(
    const bf16_t* __restrict__ A,       // AM==0
    const bf16_t* __restrict__ Abase,   // AM==1 token activations [T x K]
    const float*  __restrict__ mu,      // AM==1 mix vector [K]
    const int*    __restrict__ gidx,    // AM==1 gather list (nullable)
    const int*    __restrict__ gcntp,   // gather count (nullable)
    const float*  __restrict__ P0,      // AM==2
    const float*  __restrict__ P1,      // AM==2
    const bf16_t* __restrict__ Bw,      // [K x N]
    int M, int N, int K, int S,
    float*  __restrict__ Cf,
    bf16_t* __restrict__ Cb,
    const float* __restrict__ Xres,     // EP==3
    float*  __restrict__ Cf2,           // EP==3
    const float* __restrict__ kvbuf,    // EP==4
    const float* __restrict__ gw,       // EP==4
    float*  __restrict__ outacc) {      // EP==4
  __shared__ bf16_t sA[BM * LDSP];
  __shared__ bf16_t sB[BN * LDSP];

  const int tid = threadIdx.x;
  const int rowbase = blockIdx.y * BM;
  const int colbase = blockIdx.x * BN;

  int cnt = M;
  if (gcntp) {
    cnt = *gcntp;
    if (cnt <= 0 || rowbase >= cnt) return;
  }

  const int wid = tid >> 5, lane = tid & 31;
  const int waveM = wid >> 2;   // 0..1
  const int waveN = wid & 3;    // 0..3
  const int half = lane >> 4;
  const int l16 = lane & 15;

  v8f zero;
  #pragma unroll
  for (int i = 0; i < 8; ++i) zero[i] = 0.f;
  v8f acc[4][2];
  #pragma unroll
  for (int fm = 0; fm < 4; ++fm)
    #pragma unroll
    for (int fn = 0; fn < 2; ++fn) acc[fm][fn] = zero;

  const int nk = K / BKK;
  for (int kt = 0; kt < nk; ++kt) {
    const int k0 = kt * BKK;
    // --- stage A tile (128 x 32 bf16) ---
    for (int i = tid; i < BM * (BKK / 8); i += 256) {
      int r = i >> 2, seg = i & 3;
      int kk = k0 + seg * 8;
      union { uint4 q; bf16_t h[8]; } ld;
      if (AM == 0) {
        int g = rowbase + r;
        ld.q = *(const uint4*)(A + (size_t)g * K + kk);
      } else if (AM == 1) {
        int g = rowbase + r;
        int tok;
        if (gidx) { int gg = (g < cnt) ? g : (cnt - 1); tok = gidx[gg]; }
        else tok = g;
        int sIn = tok % S;
        const bf16_t* e0 = Abase + (size_t)tok * K + kk;
        const bf16_t* e1 = (sIn == 0) ? (const bf16_t*)0 : (Abase + (size_t)(tok - 1) * K + kk);
        #pragma unroll
        for (int j = 0; j < 8; ++j) {
          float ev = bf2f(e0[j]);
          float xv = e1 ? bf2f(e1[j]) : 0.f;
          float mm = mu[kk + j];
          ld.h[j] = f2bf(xv + mm * (ev - xv));
        }
      } else {  // AM == 2
        int g = rowbase + r;
        const float* p0 = P0 + (size_t)g * K + kk;
        const float* p1 = P1 + (size_t)g * K + kk;
        #pragma unroll
        for (int j = 0; j < 8; ++j) ld.h[j] = f2bf(p0[j] * p1[j]);
      }
      bf16_t* dst = &sA[r * LDSP + seg * 8];
      #pragma unroll
      for (int j = 0; j < 8; ++j) dst[j] = ld.h[j];
    }
    // --- stage B tile transposed: sB[n][k] = Bw[k][colbase+n] ---
    for (int i = tid; i < BN * (BKK / 8); i += 256) {
      int n = i & (BN - 1);
      int kseg = i >> 7;
      int kk = k0 + kseg * 8;
      const bf16_t* src = Bw + (size_t)kk * N + colbase + n;
      if (kt + 1 < nk) __builtin_prefetch(src + (size_t)BKK * N, 0, 1);  // global_prefetch
      bf16_t* dst = &sB[n * LDSP + kseg * 8];
      #pragma unroll
      for (int j = 0; j < 8; ++j) dst[j] = src[(size_t)j * N];
    }
    __syncthreads();

    // --- fragments + 8 WMMAs ---
    v16bf af[4], bfr[2];
    #pragma unroll
    for (int fm = 0; fm < 4; ++fm) {
      int m = waveM * 64 + fm * 16 + l16;
      const unsigned short* pr = (const unsigned short*)&sA[m * LDSP];
      union { unsigned short u[16]; v16bf v; } X;
      #pragma unroll
      for (int j = 0; j < 8; ++j) {           // ISA 16-bit A 16x32 layout
        X.u[j]     = pr[half * 8 + j];        // K = 8*half .. +7   -> VGPR0..3
        X.u[8 + j] = pr[16 + half * 8 + j];   // K = 16+8*half..+7  -> VGPR4..7
      }
      af[fm] = X.v;
    }
    #pragma unroll
    for (int fn = 0; fn < 2; ++fn) {
      int n = waveN * 32 + fn * 16 + l16;
      const unsigned short* pr = (const unsigned short*)&sB[n * LDSP];
      union { unsigned short u[16]; v16bf v; } X;
      #pragma unroll
      for (int j = 0; j < 8; ++j) {           // same K order as A: 8*half..+7, 16+8*half..+7
        X.u[j]     = pr[half * 8 + j];
        X.u[8 + j] = pr[16 + half * 8 + j];
      }
      bfr[fn] = X.v;
    }
    #pragma unroll
    for (int fm = 0; fm < 4; ++fm)
      #pragma unroll
      for (int fn = 0; fn < 2; ++fn)
        acc[fm][fn] = __builtin_amdgcn_wmma_f32_16x16x32_bf16(
            false, af[fm], false, bfr[fn], (short)0, acc[fm][fn], false, false);
    __syncthreads();
  }

  // --- epilogue ---
  #pragma unroll
  for (int fm = 0; fm < 4; ++fm) {
    #pragma unroll
    for (int fn = 0; fn < 2; ++fn) {
      int gcol = colbase + waveN * 32 + fn * 16 + l16;
      #pragma unroll
      for (int r = 0; r < 8; ++r) {
        int grow = rowbase + waveM * 64 + fm * 16 + r + 8 * half;
        float v = acc[fm][fn][r];
        if (EP == 0) {
          Cf[(size_t)grow * N + gcol] = v;
        } else if (EP == 1) {
          Cf[(size_t)grow * N + gcol] = 1.f / (1.f + __expf(-v));
        } else if (EP == 2) {
          float t = v > 0.f ? v : 0.f;
          Cb[(size_t)grow * N + gcol] = f2bf(t * t);
        } else if (EP == 3) {
          size_t o = (size_t)grow * N + gcol;
          float t = v + Xres[o];
          Cf[o] = t; Cf2[o] = t;
        } else {  // EP == 4 (MoE)
          if (grow < cnt) {
            int tok = gidx[grow];
            float wgt = gw[grow];
            float rv = 1.f / (1.f + __expf(-v));
            outacc[(size_t)tok * N + gcol] += wgt * rv * kvbuf[(size_t)grow * N + gcol];
          }
        }
      }
    }
  }
}

// ---------------------------------------------------------------------------
// Host orchestration
// ---------------------------------------------------------------------------
extern "C" void kernel_launch(void* const* d_in, const int* in_sizes, int n_in,
                              void* d_out, int out_size, void* d_ws, size_t ws_size,
                              hipStream_t stream) {
  const float* x          = (const float*)d_in[0];
  const float* ln1_g      = (const float*)d_in[1];
  const float* ln1_b      = (const float*)d_in[2];
  const float* ln2_g      = (const float*)d_in[3];
  const float* ln2_b      = (const float*)d_in[4];
  const float* tm_mu_k    = (const float*)d_in[5];
  const float* tm_mu_v    = (const float*)d_in[6];
  const float* tm_mu_r    = (const float*)d_in[7];
  const float* time_decay = (const float*)d_in[8];
  const float* time_first = (const float*)d_in[9];
  const float* Wk         = (const float*)d_in[10];
  const float* Wv         = (const float*)d_in[11];
  const float* Wr         = (const float*)d_in[12];
  const float* Wo         = (const float*)d_in[13];
  const float* router_w   = (const float*)d_in[14];
  const float* cm_mu_k    = (const float*)d_in[15];
  const float* cm_mu_r    = (const float*)d_in[16];
  const float* eWk        = (const float*)d_in[17];
  const float* eWv        = (const float*)d_in[18];
  const float* eWr        = (const float*)d_in[19];

  const int D = in_sizes[1];
  const int E = in_sizes[14] / D;
  const int F = in_sizes[17] / (E * D);
  const int S = 2048;
  const int T = in_sizes[0] / D;       // B*S
  const int Bn = T / S;
  const int L = 128, NC = S / L;

  char* w = (char*)d_ws;
  auto alloc = [&](size_t bytes) -> char* {
    char* p = w; w += (bytes + 255) & ~(size_t)255; return p;
  };
  bf16_t* xnb  = (bf16_t*)alloc((size_t)T * D * 2);
  float*  kbuf = (float*)alloc((size_t)T * D * 4);
  float*  vbuf = (float*)alloc((size_t)T * D * 4);
  float*  rbuf = (float*)alloc((size_t)T * D * 4);
  float*  wkvb = (float*)alloc((size_t)T * D * 4);
  float*  x2   = (float*)alloc((size_t)T * D * 4);
  bf16_t* eib  = (bf16_t*)alloc((size_t)T * D * 2);
  float*  Ac   = (float*)alloc((size_t)Bn * D * NC * 4);
  float*  Bc   = (float*)alloc((size_t)Bn * D * NC * 4);
  float*  Mc   = (float*)alloc((size_t)Bn * D * NC * 4);
  float*  Sa   = (float*)alloc((size_t)Bn * D * NC * 4);
  float*  Sb   = (float*)alloc((size_t)Bn * D * NC * 4);
  float*  Sp   = (float*)alloc((size_t)Bn * D * NC * 4);
  int*    counts = (int*)alloc(256);
  int*    gidx   = (int*)alloc((size_t)E * T * 4);
  float*  gww    = (float*)alloc((size_t)E * T * 4);
  bf16_t* keb  = (bf16_t*)alloc((size_t)T * F * 2);
  float*  kve  = (float*)alloc((size_t)T * D * 4);
  bf16_t* Wkb  = (bf16_t*)alloc((size_t)D * D * 2);
  bf16_t* Wvb  = (bf16_t*)alloc((size_t)D * D * 2);
  bf16_t* Wrb  = (bf16_t*)alloc((size_t)D * D * 2);
  bf16_t* Wob  = (bf16_t*)alloc((size_t)D * D * 2);
  bf16_t* eWkb = (bf16_t*)alloc((size_t)E * D * F * 2);
  bf16_t* eWvb = (bf16_t*)alloc((size_t)E * F * D * 2);
  bf16_t* eWrb = (bf16_t*)alloc((size_t)E * D * D * 2);
  (void)ws_size; (void)n_in; (void)out_size;

  float* outf = (float*)d_out;

  auto cvt = [&](const float* src, bf16_t* dst, size_t n) {
    int grid = (int)(((n + 255) / 256 > 8192) ? 8192 : (n + 255) / 256);
    cvt_f32_bf16<<<grid, 256, 0, stream>>>(src, dst, n);
  };
  cvt(Wk, Wkb, (size_t)D * D);
  cvt(Wv, Wvb, (size_t)D * D);
  cvt(Wr, Wrb, (size_t)D * D);
  cvt(Wo, Wob, (size_t)D * D);
  cvt(eWk, eWkb, (size_t)E * D * F);
  cvt(eWv, eWvb, (size_t)E * F * D);
  cvt(eWr, eWrb, (size_t)E * D * D);

  // LN1 -> xnb
  ln_kernel<<<T, 256, 0, stream>>>(x, ln1_g, ln1_b, xnb, D);

  dim3 gDD(D / BN, T / BM);   // N=D GEMMs
  dim3 gDF(F / BN, T / BM);   // N=F GEMMs

  // k = mix(xn, mu_k) @ Wk ; v = mix(..mu_v) @ Wv ; r = sigmoid(mix(..mu_r) @ Wr)
  gemm_bf16_wmma<1, 0><<<gDD, 256, 0, stream>>>(nullptr, xnb, tm_mu_k, nullptr, nullptr,
      nullptr, nullptr, Wkb, T, D, D, S, kbuf, nullptr, nullptr, nullptr, nullptr, nullptr, nullptr);
  gemm_bf16_wmma<1, 0><<<gDD, 256, 0, stream>>>(nullptr, xnb, tm_mu_v, nullptr, nullptr,
      nullptr, nullptr, Wvb, T, D, D, S, vbuf, nullptr, nullptr, nullptr, nullptr, nullptr, nullptr);
  gemm_bf16_wmma<1, 1><<<gDD, 256, 0, stream>>>(nullptr, xnb, tm_mu_r, nullptr, nullptr,
      nullptr, nullptr, Wrb, T, D, D, S, rbuf, nullptr, nullptr, nullptr, nullptr, nullptr, nullptr);

  // Chunk-parallel WKV scan
  int tot1 = Bn * D * NC;
  wkv_pass1<<<(tot1 + 255) / 256, 256, 0, stream>>>(kbuf, vbuf, time_decay, Ac, Bc, Mc, Bn, S, D, L);
  wkv_pass2<<<(Bn * D + 255) / 256, 256, 0, stream>>>(Ac, Bc, Mc, time_decay, Sa, Sb, Sp, Bn, D, NC, L);
  wkv_pass3<<<(tot1 + 255) / 256, 256, 0, stream>>>(kbuf, vbuf, time_decay, time_first,
                                                    Sa, Sb, Sp, wkvb, Bn, S, D, L);

  // x2 = x + (r*wkv) @ Wo   (store to ws x2 and to d_out)
  gemm_bf16_wmma<2, 3><<<gDD, 256, 0, stream>>>(nullptr, nullptr, nullptr, nullptr, nullptr,
      rbuf, wkvb, Wob, T, D, D, S, x2, nullptr, x, outf, nullptr, nullptr, nullptr);

  // LN2 -> eib ; router -> top-2 gather lists
  ln_kernel<<<T, 256, 0, stream>>>(x2, ln2_g, ln2_b, eib, D);
  hipMemsetAsync(counts, 0, E * sizeof(int), stream);
  router_kernel<<<T, 256, 0, stream>>>(eib, router_w, D, T, counts, gidx, gww);

  // MoE: grouped GEMMs over gathered (routed) rows only
  for (int e = 0; e < E; ++e) {
    const int* gi = gidx + (size_t)e * T;
    const float* gwp = gww + (size_t)e * T;
    const int* cp = counts + e;
    // k_e = relu(mix(ei, cm_mu_k[e]) @ eWk[e])^2  -> bf16
    gemm_bf16_wmma<1, 2><<<gDF, 256, 0, stream>>>(nullptr, eib, cm_mu_k + (size_t)e * D, gi, cp,
        nullptr, nullptr, eWkb + (size_t)e * D * F, T, F, D, S,
        nullptr, keb, nullptr, nullptr, nullptr, nullptr, nullptr);
    // kv_e = k_e @ eWv[e] -> f32
    gemm_bf16_wmma<0, 0><<<gDD, 256, 0, stream>>>(keb, nullptr, nullptr, nullptr, cp,
        nullptr, nullptr, eWvb + (size_t)e * F * D, T, D, F, S,
        kve, nullptr, nullptr, nullptr, nullptr, nullptr, nullptr);
    // out += gw * sigmoid(mix(ei, cm_mu_r[e]) @ eWr[e]) * kv_e   (scatter by token)
    gemm_bf16_wmma<1, 4><<<gDD, 256, 0, stream>>>(nullptr, eib, cm_mu_r + (size_t)e * D, gi, cp,
        nullptr, nullptr, eWrb + (size_t)e * D * D, T, D, D, S,
        nullptr, nullptr, nullptr, nullptr, kve, gwp, outf);
  }
}
